// TopKGate_89043261980984
// MI455X (gfx1250) — compile-verified, matrix operand order checked by reading, not run on hardware
//
#include <hip/hip_runtime.h>
#include <hip/hip_bf16.h>

#define S       4096
#define E       64
#define MD      1024
#define CAP     128
#define TOPK    4
#define THRESH  0.2f
#define SEC     ((size_t)S * E * CAP)   // 33,554,432

typedef float v2f  __attribute__((ext_vector_type(2)));
typedef float v8f  __attribute__((ext_vector_type(8)));
typedef __bf16 v16bf __attribute__((ext_vector_type(16)));

// ---------------------------------------------------------------------------
// Kernel 1: logits = x @ wg^T via fp32 WMMA (16x16x4), one wave per 16x16 tile
// ---------------------------------------------------------------------------
#if __has_builtin(__builtin_amdgcn_wmma_f32_16x16x4_f32)
__global__ __launch_bounds__(32) void gemm_logits(const float* __restrict__ x,
                                                  const float* __restrict__ wg,
                                                  float* __restrict__ logits) {
  const int lane = threadIdx.x;
  const int half = lane >> 4;          // 0/1 -> K sub-pair
  const int r    = lane & 15;          // row (A: M, B: N)
  const int m0 = blockIdx.x * 16;
  const int n0 = blockIdx.y * 16;
  const float* ax = x  + (size_t)(m0 + r) * MD + 2 * half;  // A[M=r][K=2h,2h+1]
  const float* bw = wg + (size_t)(n0 + r) * MD + 2 * half;  // B[K][N=r] = wg[N][K]
  v8f c = {};
#pragma unroll 4
  for (int k = 0; k < MD; k += 4) {
    v2f a = *(const v2f*)(ax + k);
    v2f b = *(const v2f*)(bw + k);
    c = __builtin_amdgcn_wmma_f32_16x16x4_f32(false, a, false, b, (short)0, c,
                                              false, false);
  }
#pragma unroll
  for (int v = 0; v < 8; ++v) {
    int m = m0 + v + 8 * half;         // C/D layout: VGPR v -> M=v (+8 for hi half)
    logits[(size_t)m * E + n0 + r] = c[v];
  }
}
#else
// Fallback: bf16 WMMA 16x16x32 (codegen-confirmed builtin)
__global__ __launch_bounds__(32) void gemm_logits(const float* __restrict__ x,
                                                  const float* __restrict__ wg,
                                                  float* __restrict__ logits) {
  const int lane = threadIdx.x;
  const int half = lane >> 4;
  const int r    = lane & 15;
  const int m0 = blockIdx.x * 16;
  const int n0 = blockIdx.y * 16;
  const float* ax = x  + (size_t)(m0 + r) * MD;
  const float* bw = wg + (size_t)(n0 + r) * MD;
  v8f c = {};
  for (int k0 = 0; k0 < MD; k0 += 32) {
    v16bf a, b;
#pragma unroll
    for (int j = 0; j < 8; ++j) {
      int kb = k0 + (j & 3) * 2 + 8 * half + 16 * (j >> 2);
      a[2*j]   = (__bf16)ax[kb];
      a[2*j+1] = (__bf16)ax[kb + 1];
      b[2*j]   = (__bf16)bw[kb];
      b[2*j+1] = (__bf16)bw[kb + 1];
    }
    c = __builtin_amdgcn_wmma_f32_16x16x32_bf16(false, a, false, b, (short)0, c,
                                                false, false);
  }
#pragma unroll
  for (int v = 0; v < 8; ++v) {
    int m = m0 + v + 8 * half;
    logits[(size_t)m * E + n0 + r] = c[v];
  }
}
#endif

// ---------------------------------------------------------------------------
// Kernel 0: zero the atomic accumulators
// ---------------------------------------------------------------------------
__global__ void zero_stats(int* cnt1, int* row_has, float* me_sum) {
  int i = threadIdx.x;
  if (i < E) { cnt1[i] = 0; me_sum[i] = 0.f; }
  if (i < TOPK) row_has[i] = 0;
}

// ---------------------------------------------------------------------------
// Kernel 2: per-token gating — softmax, argmax, top-4 + threshold cut.
// One wave per token (wave32: each lane owns experts {lane, lane+32}).
// ---------------------------------------------------------------------------
__global__ __launch_bounds__(256) void gate_kernel(const float* __restrict__ logits,
                                                   float* __restrict__ gates,
                                                   int* __restrict__ chosen,
                                                   int* __restrict__ cnt1,
                                                   int* __restrict__ row_has) {
  const int wave = threadIdx.x >> 5;
  const int lane = threadIdx.x & 31;
  const int s = blockIdx.x * 8 + wave;
  const float* lg = logits + (size_t)s * E;
  float g0 = lg[lane], g1 = lg[lane + 32];

  float m = fmaxf(g0, g1);
  for (int off = 16; off; off >>= 1) m = fmaxf(m, __shfl_xor(m, off, 32));
  float e0 = __expf(g0 - m), e1 = __expf(g1 - m);
  float ssum = e0 + e1;
  for (int off = 16; off; off >>= 1) ssum += __shfl_xor(ssum, off, 32);
  float inv = 1.0f / ssum;
  g0 = e0 * inv; g1 = e1 * inv;
  gates[(size_t)s * E + lane]      = g0;
  gates[(size_t)s * E + lane + 32] = g1;

  // argmax (first-index tie-break) -> mask1 counts
  float av = g0; int ai = lane;
  if (g1 > av) { av = g1; ai = lane + 32; }
  for (int off = 16; off; off >>= 1) {
    float ov = __shfl_xor(av, off, 32); int oi = __shfl_xor(ai, off, 32);
    if (ov > av || (ov == av && oi < ai)) { av = ov; ai = oi; }
  }
  if (lane == 0) atomicAdd(&cnt1[ai], 1);

  // top-4 by repeated masked max (ties -> lower index, matching lax.top_k)
  float w0 = g0, w1 = g1;
  float tv[TOPK]; int ti[TOPK];
#pragma unroll
  for (int t = 0; t < TOPK; ++t) {
    float v = w0; int vi = lane;
    if (w1 > v) { v = w1; vi = lane + 32; }
    for (int off = 16; off; off >>= 1) {
      float ov = __shfl_xor(v, off, 32); int oi = __shfl_xor(vi, off, 32);
      if (ov > v || (ov == v && oi < vi)) { v = ov; vi = oi; }
    }
    tv[t] = v; ti[t] = vi;
    if (vi == lane) w0 = -1e30f;
    else if (vi == lane + 32) w1 = -1e30f;
  }

  if (lane == 0) {
    float pre = 0.f;
#pragma unroll
    for (int t = 0; t < TOPK; ++t) {
      bool flag = (t == 0) || (pre < THRESH);   // (cum - g_sorted) < 0.2, slot0 forced
      pre += tv[t];
      chosen[t * S + s] = flag ? (ti[t] + 1) : 0;
      if (flag) atomicOr(&row_has[t], 1);
    }
  }
}

// ---------------------------------------------------------------------------
// Kernel 3: me[e] = column sums of gates (for l_aux)
// ---------------------------------------------------------------------------
__global__ __launch_bounds__(256) void colreduce(const float* __restrict__ gates,
                                                 float* __restrict__ me_sum) {
  __shared__ float sm[256];
  const int e = blockIdx.x;
  float acc = 0.f;
  for (int s = threadIdx.x; s < S; s += 256) acc += gates[(size_t)s * E + e];
  sm[threadIdx.x] = acc; __syncthreads();
  for (int off = 128; off; off >>= 1) {
    if (threadIdx.x < off) sm[threadIdx.x] += sm[threadIdx.x + off];
    __syncthreads();
  }
  if (threadIdx.x == 0) me_sum[e] = sm[0];
}

// ---------------------------------------------------------------------------
// Kernel 4: capacity enforcement — per-expert prefix scan over flat (t,s) rows;
// keep first CAP ones; loc[r] = position or -1. One wave per expert.
// ---------------------------------------------------------------------------
__global__ __launch_bounds__(32) void capacity_kernel(const int* __restrict__ chosen,
                                                      const int* __restrict__ row_has,
                                                      int* __restrict__ loc) {
  const int e = blockIdx.x;
  const int lane = threadIdx.x;
  const int nnz = row_has[0] + row_has[1] + row_has[2] + row_has[3];
  const int stop = nnz < TOPK ? nnz : TOPK;   // stop_index (cap = int(2*cf) = 4)
  int running = 0;
  for (int base = 0; base < TOPK * S; base += 32) {
    const int r = base + lane;
    const int t = r >> 12;                    // r / S
    int flag = (chosen[r] == e + 1 && t < stop) ? 1 : 0;
    int xv = flag;
    for (int off = 1; off < 32; off <<= 1) {  // inclusive scan
      int y = __shfl_up(xv, off, 32);
      if (lane >= off) xv += y;
    }
    int myloc = running + xv - 1;
    loc[r] = (flag && myloc < CAP) ? myloc : -1;
    running += __shfl(xv, 31, 32);
  }
}

// ---------------------------------------------------------------------------
// Kernel 5: zero-fill the whole output (HBM-store bound part), float4 stores
// ---------------------------------------------------------------------------
__global__ void zerofill(float* __restrict__ out, size_t n) {
  size_t i = ((size_t)blockIdx.x * blockDim.x + threadIdx.x) * 4;
  if (i + 4 <= n) {
    *(float4*)(out + i) = make_float4(0.f, 0.f, 0.f, 0.f);
  } else {
    for (; i < n; ++i) out[i] = 0.f;
  }
}

// ---------------------------------------------------------------------------
// Kernel 6: scatter kept entries into combine_weights / dispatch_mask
// ---------------------------------------------------------------------------
__global__ void scatter_kernel(const int* __restrict__ chosen,
                               const int* __restrict__ loc,
                               const float* __restrict__ gates,
                               float* __restrict__ out) {
  const int r = blockIdx.x * blockDim.x + threadIdx.x;
  if (r >= TOPK * S) return;
  const int l = loc[r];
  if (l < 0) return;
  const int s = r & (S - 1);
  const int e = chosen[r] - 1;
  const float val = gates[(size_t)s * E + e];
  const size_t off = 1 + ((size_t)(s * E + e)) * CAP + l;
  out[off] = val;                // combine_weights[s,e,l]
  out[off + SEC] = 1.0f;         // dispatch_mask (val = softmax > 0)
}

// ---------------------------------------------------------------------------
// Kernel 7: l_aux and exp_counts
// ---------------------------------------------------------------------------
__global__ __launch_bounds__(64) void finalize(const float* __restrict__ me_sum,
                                               const int* __restrict__ cnt1,
                                               float* __restrict__ out) {
  __shared__ float sm[64];
  const int e = threadIdx.x;
  sm[e] = (me_sum[e] / (float)S) * ((float)cnt1[e] / (float)S);
  __syncthreads();
  for (int off = 32; off; off >>= 1) {
    if (e < off) sm[e] += sm[e + off];
    __syncthreads();
  }
  if (e == 0) out[0] = sm[0] * (float)E;             // l_aux
  out[1 + 2 * SEC + e] = (float)cnt1[e];             // exp_counts
}

// ---------------------------------------------------------------------------
extern "C" void kernel_launch(void* const* d_in, const int* in_sizes, int n_in,
                              void* d_out, int out_size, void* d_ws, size_t ws_size,
                              hipStream_t stream) {
  const float* x  = (const float*)d_in[0];   // [S, MD]
  const float* wg = (const float*)d_in[1];   // [E, MD]
  float* out = (float*)d_out;

  // workspace layout (~2.3 MB)
  float* logits = (float*)d_ws;              // S*E
  float* gates  = logits + (size_t)S * E;    // S*E
  int*   chosen = (int*)(gates + (size_t)S * E);  // TOPK*S
  int*   loc    = chosen + TOPK * S;              // TOPK*S
  float* me_sum = (float*)(loc + TOPK * S);       // E
  int*   cnt1   = (int*)(me_sum + E);             // E
  int*   row_has= cnt1 + E;                       // TOPK

  zero_stats<<<1, 128, 0, stream>>>(cnt1, row_has, me_sum);

  gemm_logits<<<dim3(S / 16, E / 16), 32, 0, stream>>>(x, wg, logits);

  gate_kernel<<<S / 8, 256, 0, stream>>>(logits, gates, chosen, cnt1, row_has);

  colreduce<<<E, 256, 0, stream>>>(gates, me_sum);

  capacity_kernel<<<E, 32, 0, stream>>>(chosen, row_has, loc);

  const size_t n = (size_t)out_size;
  const int zf_grid = (int)(((n + 3) / 4 + 255) / 256);
  zerofill<<<zf_grid, 256, 0, stream>>>(out, n);

  scatter_kernel<<<(TOPK * S + 255) / 256, 256, 0, stream>>>(chosen, loc, gates, out);

  finalize<<<1, 64, 0, stream>>>(me_sum, cnt1, out);
}
